// MomentumAttention_22265110463151
// MI455X (gfx1250) — matrix operand order repaired and checked by reading, not compiled, not run on hardware
//
#include <hip/hip_runtime.h>
#include <cstdint>

typedef __attribute__((ext_vector_type(16))) __bf16         v16bf;
typedef __attribute__((ext_vector_type(16))) unsigned short v16us;
typedef __attribute__((ext_vector_type(8)))  float          v8f;
typedef __attribute__((ext_vector_type(4)))  unsigned int   v4u;
typedef __attribute__((ext_vector_type(8)))  int            v8i32;
typedef __attribute__((ext_vector_type(4)))  int            v4i32;

namespace {
constexpr int   Bdim   = 2, Hdim = 16, Ldim = 8192, Ddim = 128;
constexpr int   BH     = Bdim * Hdim;
constexpr float SCALE  = 0.10511205190671433f;   // 8192^(-1/4)
constexpr float LOG_MU = -0.10536051565782630f;  // ln(0.9)
constexpr float MW_COEF = 10.0f;                 // STEPSIZE / (1 - MU)
constexpr int   NSPLIT = 16;
constexpr int   CHUNK  = 32;
constexpr int   ROWS_PER_SPLIT = Ldim / NSPLIT;  // 512
}

__device__ __forceinline__ unsigned short f2bf(float x) {
  __bf16 h = (__bf16)x;                       // native f32->bf16 cvt
  return __builtin_bit_cast(unsigned short, h);
}
__device__ __forceinline__ unsigned int pack2bf(float lo, float hi) {
  return (unsigned int)f2bf(lo) | ((unsigned int)f2bf(hi) << 16);
}
__device__ __forceinline__ float elu1(float x) {  // elu(x) + 1
  return x > 0.f ? x + 1.f : __expf(x);
}

// gfx1250 async copy global -> LDS (tracked by ASYNCcnt)
__device__ __forceinline__ void async_b128(uint32_t lds, const float* g) {
  asm volatile("global_load_async_to_lds_b128 %0, %1, off"
               :: "v"(lds), "v"(g) : "memory");
}
__device__ __forceinline__ void wait_async0() {
  asm volatile("s_wait_asynccnt 0" ::: "memory");
}
__device__ __forceinline__ uint32_t lds_off(const void* p) {
  return (uint32_t)(uintptr_t)p;  // low 32 bits of flat shared ptr == LDS offset
}

// Tensor Data Mover: DMA a 2D tile (f32 elements) global -> LDS, TENSORcnt.
// D# packing per CDNA5 ISA.
// This toolchain's builtin: 6 args (g0 u32x4, g1 i32x8, g2 i32x4, g3 i32x4,
// extra i32x8, cpol i32).
__device__ __forceinline__ void tdm_load_2d(uint32_t lds_addr, const void* gptr,
                                            uint32_t tile_w, uint32_t tile_h,
                                            uint32_t tensor_w, uint32_t tensor_h,
                                            uint32_t stride_elems) {
  uint64_t ga = (uint64_t)(uintptr_t)gptr;
  v4u g0;
  g0[0] = 1u;                                   // count=1, user descriptor
  g0[1] = lds_addr;                             // lds_addr [63:32]
  g0[2] = (unsigned int)ga;                     // global_addr [95:64]
  g0[3] = (unsigned int)((ga >> 32) & 0x1FFFFFFu) | (2u << 30);  // type=2
  v8i32 g1;
  g1[0] = (int)(2u << 16);                      // wg_mask=0, data_size=4B
  g1[1] = (int)((tensor_w & 0xFFFFu) << 16);    // tensor_dim0[15:0] @bits63:48
  g1[2] = (int)((tensor_w >> 16) | ((tensor_h & 0xFFFFu) << 16)); // td0 hi | td1 lo
  g1[3] = (int)((tensor_h >> 16) | (tile_w << 16));               // td1 hi | tile_dim0
  g1[4] = (int)(tile_h & 0xFFFFu);              // tile_dim1 (tile_dim2 = 0)
  g1[5] = (int)stride_elems;                    // tensor_dim0_stride[31:0]
  g1[6] = 0;                                    // stride0[47:32] | stride1[15:0]
  g1[7] = 0;                                    // stride1[47:16]
  v4i32 z4 = {0, 0, 0, 0};
  v8i32 z8 = {0, 0, 0, 0, 0, 0, 0, 0};
  __builtin_amdgcn_tensor_load_to_lds(g0, g1, z4, z4, z8, 0);
}
__device__ __forceinline__ void wait_tensor0() {
  __builtin_amdgcn_s_wait_tensorcnt(0);
}

__global__ void zero_ws_kernel(float* p, int n) {
  int i = blockIdx.x * blockDim.x + threadIdx.x;
  if (i < n) p[i] = 0.f;
}

// ---------------------------------------------------------------------------
// Phase 1: KV[bh][d][e] = sum_l Kf[l,d] * Vf[l,e]   (bf16 WMMA, f32 accum)
// ---------------------------------------------------------------------------
__global__ void __launch_bounds__(256)
mom_attn_kv(const float* __restrict__ K, const float* __restrict__ V,
            const float* __restrict__ mask, float* __restrict__ kv_ws) {
  __shared__ float        sK32[2][CHUNK * Ddim];   // 2 x 16 KB (TDM dst)
  __shared__ float        sV32[2][CHUNK * Ddim];   // 2 x 16 KB (TDM dst)
  __shared__ unsigned int sVb[8 * 32 * 8];         // bf16 B frags [t][lane][8u32]
  __shared__ float        sKm[2][CHUNK], sVm[2][CHUNK];

  const int bh    = blockIdx.x;
  const int b     = bh / Hdim;
  const int split = blockIdx.y;
  const int tid   = threadIdx.x;
  const int wv    = tid >> 5;        // wave id 0..7 -> d-strip
  const int lane  = tid & 31;
  const int hi    = lane >> 4;
  const int mrow  = lane & 15;

  v8f acc[8] = {};
  const size_t base_bh = (size_t)bh * Ldim * Ddim;
  const int    lbase   = split * ROWS_PER_SPLIT;

  auto issue_tdm = [&](int buf, int l0) {   // wave 0 only
    tdm_load_2d(lds_off(&sK32[buf][0]), K + base_bh + (size_t)l0 * Ddim,
                /*tile_w*/Ddim, /*tile_h*/CHUNK, /*tensor_w*/Ddim,
                /*tensor_h*/Ldim - l0, /*stride*/Ddim);
    tdm_load_2d(lds_off(&sV32[buf][0]), V + base_bh + (size_t)l0 * Ddim,
                Ddim, CHUNK, Ddim, Ldim - l0, Ddim);
  };
  auto stage_mask = [&](int mb, int l0) {
    if (tid < CHUNK) {
      int   l = l0 + tid;
      float m = mask[(size_t)b * Ldim + l];
      float w = __expf((float)(Ldim - l) * LOG_MU);   // 0.9^(L-l)
      sKm[mb][tid] = m * SCALE * (MW_COEF * (1.f - w));
      sVm[mb][tid] = m;
    }
  };

  constexpr int NCH = ROWS_PER_SPLIT / CHUNK;   // 16
  if (wv == 0) issue_tdm(0, lbase);
  stage_mask(0, lbase);

  for (int c = 0; c < NCH; ++c) {
    const int buf = c & 1;
    if (wv == 0) wait_tensor0();    // f32 buf[c&1] landed
    __syncthreads();

    // ---- cooperative V -> bf16 fragment-layout conversion (each elem once) --
    {
      const int t   = tid >> 5;
      const int ln  = tid & 31;
      const int col = t * 16 + (ln & 15);
      const int kb  = (ln >> 4) * 16;
      unsigned int u[8];
#pragma unroll
      for (int j = 0; j < 8; ++j) {
        int k0 = kb + 2 * j;
        u[j] = pack2bf(sV32[buf][k0 * Ddim + col]       * sVm[buf][k0],
                       sV32[buf][(k0 + 1) * Ddim + col] * sVm[buf][k0 + 1]);
      }
      uint4* dst = (uint4*)&sVb[tid * 8];
      dst[0] = make_uint4(u[0], u[1], u[2], u[3]);
      dst[1] = make_uint4(u[4], u[5], u[6], u[7]);
    }

    // ---- A fragment (Kf^T 16x32) in registers, per-wave unique -------------
    v16us au;
#pragma unroll
    for (int e = 0; e < 16; ++e) {
      int k = e + 16 * hi;  // contiguous k order (matches B frags)
      au[e] = f2bf(elu1(sK32[buf][k * Ddim + wv * 16 + mrow]) * sKm[buf][k]);
    }
    v16bf a = __builtin_bit_cast(v16bf, au);

    if (c + 1 < NCH) stage_mask((c + 1) & 1, lbase + (c + 1) * CHUNK);
    __syncthreads();        // sVb ready; f32 buf consumed

    if (c + 1 < NCH && wv == 0)
      issue_tdm(1 - buf, lbase + (c + 1) * CHUNK);  // DMA overlaps WMMA phase

#pragma unroll
    for (int t = 0; t < 8; ++t) {
      v16us bu = *(const v16us*)&sVb[(t * 32 + lane) * 8];
      v16bf bfrag = __builtin_bit_cast(v16bf, bu);
      acc[t] = __builtin_amdgcn_wmma_f32_16x16x32_bf16(
          false, a, false, bfrag, (short)0, acc[t], false, false);
    }
  }

  float* dst = kv_ws + (size_t)bh * Ddim * Ddim;
#pragma unroll
  for (int t = 0; t < 8; ++t) {
#pragma unroll
    for (int r = 0; r < 8; ++r) {
      int row = wv * 16 + r + 8 * hi;     // C layout: VGPR r -> M = r + 8*hi
      int col = t * 16 + mrow;
      unsafeAtomicAdd(&dst[row * Ddim + col], acc[t][r]);  // global_atomic_add_f32
    }
  }
}

// ---------------------------------------------------------------------------
// Phase 2: X[l,e] = sum_d Qf[l,d] * KV[d,e]
// ---------------------------------------------------------------------------
__global__ void __launch_bounds__(256)
mom_attn_x(const float* __restrict__ Q, const float* __restrict__ kv_ws,
           float* __restrict__ X) {
  __shared__ float        sQ32[2][128 * CHUNK];    // 2 x 16 KB (TDM dst)
  __shared__ float        sKV32[2][CHUNK * Ddim];  // 2 x 16 KB (async dst)
  __shared__ unsigned int sKVb[8 * 32 * 8];        // bf16 B frags, 8 KB

  const int bh   = blockIdx.x;
  const int row0 = blockIdx.y * 128;
  const int tid  = threadIdx.x;
  const int wv   = tid >> 5;
  const int lane = tid & 31;
  const int hi   = lane >> 4;
  const int mrow = lane & 15;

  v8f acc[8] = {};
  const float* kvb = kv_ws + (size_t)bh * Ddim * Ddim;

  auto issue = [&](int buf, int k0) {
    // Q tile: 128 rows x 32 cols, strided -> one TDM descriptor (wave 0)
    if (wv == 0)
      tdm_load_2d(lds_off(&sQ32[buf][0]),
                  Q + ((size_t)bh * Ldim + row0) * Ddim + k0,
                  /*tile_w*/CHUNK, /*tile_h*/128, /*tensor_w*/Ddim,
                  /*tensor_h*/Ldim - row0, /*stride*/Ddim);
    // KV chunk: contiguous 16 KB -> async b128 per lane (all waves)
#pragma unroll
    for (int j = 0; j < 4; ++j) {
      int i = (j * 256 + tid) * 4;
      async_b128(lds_off(&sKV32[buf][i]), kvb + (size_t)k0 * Ddim + i);
    }
  };

  constexpr int NCH = Ddim / CHUNK;   // 4
  issue(0, 0);

  for (int kc = 0; kc < NCH; ++kc) {
    const int buf = kc & 1;
    if (wv == 0) wait_tensor0();
    wait_async0();
    __syncthreads();

    // ---- cooperative KV -> bf16 fragment-layout conversion -----------------
    {
      const int t   = tid >> 5;
      const int ln  = tid & 31;
      const int col = t * 16 + (ln & 15);
      const int kb  = (ln >> 4) * 16;
      unsigned int u[8];
#pragma unroll
      for (int j = 0; j < 8; ++j) {
        int k0 = kb + 2 * j;
        u[j] = pack2bf(sKV32[buf][k0 * Ddim + col],
                       sKV32[buf][(k0 + 1) * Ddim + col]);
      }
      uint4* dst = (uint4*)&sKVb[tid * 8];
      dst[0] = make_uint4(u[0], u[1], u[2], u[3]);
      dst[1] = make_uint4(u[4], u[5], u[6], u[7]);
    }

    // ---- A fragment (Qf 16x32) in registers --------------------------------
    v16us au;
#pragma unroll
    for (int e = 0; e < 16; ++e) {
      int k = e + 16 * hi;
      au[e] = f2bf(elu1(sQ32[buf][(wv * 16 + mrow) * CHUNK + k]) * SCALE);
    }
    v16bf a = __builtin_bit_cast(v16bf, au);

    __syncthreads();
    if (kc + 1 < NCH) issue(1 - buf, (kc + 1) * CHUNK);

#pragma unroll
    for (int t = 0; t < 8; ++t) {
      v16us bu = *(const v16us*)&sKVb[(t * 32 + lane) * 8];
      v16bf bfrag = __builtin_bit_cast(v16bf, bu);
      acc[t] = __builtin_amdgcn_wmma_f32_16x16x32_bf16(
          false, a, false, bfrag, (short)0, acc[t], false, false);
    }
  }

#pragma unroll
  for (int t = 0; t < 8; ++t) {
#pragma unroll
    for (int r = 0; r < 8; ++r) {
      int row = row0 + wv * 16 + r + 8 * hi;
      int col = t * 16 + mrow;
      X[((size_t)bh * Ldim + row) * Ddim + col] = acc[t][r];
    }
  }
}

extern "C" void kernel_launch(void* const* d_in, const int* in_sizes, int n_in,
                              void* d_out, int out_size, void* d_ws, size_t ws_size,
                              hipStream_t stream) {
  (void)in_sizes; (void)n_in; (void)out_size; (void)ws_size;
  const float* Q    = (const float*)d_in[0];
  const float* K    = (const float*)d_in[1];
  const float* V    = (const float*)d_in[2];
  const float* mask = (const float*)d_in[3];
  float* X     = (float*)d_out;
  float* kv_ws = (float*)d_ws;           // BH * 128 * 128 f32 = 2 MB

  const int kvn = BH * Ddim * Ddim;
  zero_ws_kernel<<<(kvn + 255) / 256, 256, 0, stream>>>(kv_ws, kvn);
  mom_attn_kv<<<dim3(BH, NSPLIT), 256, 0, stream>>>(K, V, mask, kv_ws);
  mom_attn_x<<<dim3(BH, Ldim / 128), 256, 0, stream>>>(Q, kv_ws, X);
}
